// BlockSparseAttention_kernel_18133351923798
// MI455X (gfx1250) — matrix operand order repaired and checked by reading, not compiled, not run on hardware
//
#include <hip/hip_runtime.h>

// Block-sparse causal flash attention for gfx1250 (MI455X).
// B=2, T=2048, H=16, Dqk=Dv=128, BLOCK_M=BLOCK_N=64, causal, fp16 I/O.
// One workgroup per (q-block, head, batch); 4 waves x 32 lanes.
// v_wmma_f32_16x16x32_f16 for QK^T and PV; K/V staged by the Tensor Data
// Mover (tensor_load_to_lds, TENSORcnt) with K double-buffered so the next
// kv block's DMA overlaps the current block's compute.

typedef __attribute__((ext_vector_type(16))) _Float16     v16h;
typedef __attribute__((ext_vector_type(8)))  _Float16     v8h;
typedef __attribute__((ext_vector_type(8)))  float        v8f;
typedef __attribute__((ext_vector_type(4)))  unsigned int u32x4;
typedef __attribute__((ext_vector_type(8)))  int          i32x8;
typedef __attribute__((ext_vector_type(4)))  int          i32x4;

#define T_DIM   2048
#define H_DIM   16
#define D_DIM   128
#define BLK     64
#define NQ      (T_DIM / BLK)

// LDS strides in halves (+8 pad vs 64 banks)
#define KSTR    136   // K / raw-V tiles: 64 rows x 128 (+8) — matches TDM pad
#define VSTR    72    // V^T rows: 128 x 64 (+8)
#define PSTR    72    // P tiles: 4*16 x 64 (+8)

__device__ __forceinline__ v16h load_frag(const _Float16* lo, const _Float16* hi) {
  v8h a = *(const v8h*)lo;
  v8h b = *(const v8h*)hi;
  v16h r;
#pragma unroll
  for (int i = 0; i < 8; ++i) { r[i] = a[i]; r[i + 8] = b[i]; }
  return r;
}

// TDM: DMA one 64x128 fp16 tile (row stride H*D elements) into LDS,
// padding LDS dest by 4 DWORDs (8 halves) after every 64 DWORDs (128 halves)
// -> LDS row stride = 136 halves, exactly KSTR.
__device__ __forceinline__ void tdm_load_tile(unsigned int lds_addr,
                                              const _Float16* gptr) {
  unsigned long long ga = (unsigned long long)(uintptr_t)gptr;
  u32x4 g0;
  g0[0] = 1u;                                   // count=1, user descriptor
  g0[1] = lds_addr;                             // LDS byte address
  g0[2] = (unsigned int)ga;                     // global_addr[31:0]
  g0[3] = ((unsigned int)(ga >> 32) & 0x01FFFFFFu) | (2u << 30); // [56:32], type=2
  i32x8 g1;
  g1[0] = (1 << 16)          // data_size = 2 bytes
        | (1 << 20)          // pad_enable
        | (5 << 22)          // pad_interval: 64 DWORDs (= one 128-half row)
        | (3 << 25);         // pad_amount: 4 DWORDs (= 8 halves)
  g1[1] = (int)(128u << 16); // tensor_dim0 = 128 (bits [79:48], low half)
  g1[2] = (int)(64u << 16);  // tensor_dim1 = 64  (bits [111:80], low half)
  g1[3] = (int)(128u << 16); // tile_dim0 = 128   (bits [127:112])
  g1[4] = 64;                // tile_dim1 = 64, tile_dim2 = 0
  g1[5] = H_DIM * D_DIM;     // tensor_dim0_stride = 2048 elements
  g1[6] = 0;
  g1[7] = 0;
  i32x4 z4 = {0, 0, 0, 0};           // groups 2/3 unused for 2-D tensor
  i32x8 z8 = {0, 0, 0, 0, 0, 0, 0, 0};
  __builtin_amdgcn_tensor_load_to_lds(g0, g1, z4, z4, z8, 0);
}

__global__ __launch_bounds__(128, 1)
void bsattn_fwd_kernel(const _Float16* __restrict__ q,
                       const _Float16* __restrict__ k,
                       const _Float16* __restrict__ v,
                       const unsigned char* __restrict__ bmask,
                       _Float16* __restrict__ out)
{
  __shared__ _Float16 ldsK[2][BLK * KSTR];   // 2 x 17408 B (double buffer)
  __shared__ _Float16 ldsVr[BLK * KSTR];     // 17408 B raw V (row-major)
  __shared__ _Float16 ldsVt[D_DIM * VSTR];   // 18432 B transposed V
  __shared__ _Float16 ldsP[4 * 16 * PSTR];   //  9216 B P tiles

  const int tid    = threadIdx.x;
  const int w      = tid >> 5;
  const int lane   = tid & 31;
  const int ln     = lane & 15;
  const int hi     = lane >> 4;
  const int koff8  = hi * 8;
  const int koff16 = hi * 16;

  const int qb = blockIdx.x;
  const int h  = blockIdx.y;
  const int b  = blockIdx.z;

  const size_t bh_off = (((size_t)b * T_DIM) * H_DIM + h) * (size_t)D_DIM;
  const _Float16* kbh = k + bh_off;   // + t*H*D for token t
  const _Float16* vbh = v + bh_off;

  // ---- Q fragments (A-matrix 16x32 f16 layout), resident in registers ----
  const size_t qrow_off =
      ((((size_t)b * T_DIM) + (size_t)(qb * BLK + w * 16 + ln)) * H_DIM + h) * D_DIM;
  v16h qf[4];
#pragma unroll
  for (int kc = 0; kc < 4; ++kc) {
    const _Float16* p = q + qrow_off + kc * 32 + koff8;
    qf[kc] = load_frag(p, p + 16);
  }

  float m_i[8], l_i[8];
  v8f zero = {};
  v8f o[8];
#pragma unroll
  for (int r = 0; r < 8; ++r) { m_i[r] = -1e30f; l_i[r] = 0.0f; }
#pragma unroll
  for (int j = 0; j < 8; ++j) o[j] = zero;

  const unsigned char* bm = bmask + ((size_t)(b * H_DIM + h) * NQ + qb) * NQ;
  const float scale = 0.08838834764831845f;  // 1/sqrt(128)

  // ---- pipeline prologue: find first active kv block, kick off its DMA ----
  int cur = -1;
  for (int kb = 0; kb <= qb; ++kb) if (bm[kb]) { cur = kb; break; }
  int p = 0;
  if (cur >= 0 && w == 0) {
    tdm_load_tile((unsigned int)(uintptr_t)&ldsK[0][0],
                  kbh + (size_t)(cur * BLK) * (H_DIM * D_DIM));
    tdm_load_tile((unsigned int)(uintptr_t)&ldsVr[0],
                  vbh + (size_t)(cur * BLK) * (H_DIM * D_DIM));
  }

  while (cur >= 0) {
    const bool diag = (cur == qb);
    int nxt = -1;
    for (int kb = cur + 1; kb <= qb; ++kb) if (bm[kb]) { nxt = kb; break; }

    // ---- current tiles landed in LDS ----
    if (w == 0) __builtin_amdgcn_s_wait_tensorcnt((short)0);
    __syncthreads();                               // B1: K[p], Vr valid

    // ---- transpose raw V -> Vt (packed b32 stores); frees ldsVr ----
#pragma unroll
    for (int i = 0; i < 4; ++i) {
      int pi  = tid + i * 128;                     // 512 tasks: 32 row-pairs x 16 segs
      int rp  = pi >> 4;
      int seg = pi & 15;
      v8h a0 = *(const v8h*)&ldsVr[(2 * rp)     * KSTR + seg * 8];
      v8h a1 = *(const v8h*)&ldsVr[(2 * rp + 1) * KSTR + seg * 8];
#pragma unroll
      for (int d8 = 0; d8 < 8; ++d8) {
        _Float16 x = a0[d8], y = a1[d8];
        unsigned int pk = (unsigned int)__builtin_bit_cast(unsigned short, x)
                        | ((unsigned int)__builtin_bit_cast(unsigned short, y) << 16);
        *(unsigned int*)&ldsVt[(seg * 8 + d8) * VSTR + 2 * rp] = pk;
      }
    }
    __syncthreads();                               // B2: Vt valid, ldsVr free

    // ---- overlap: DMA next block's K (other buffer) and V (freed raw buf) ----
    if (nxt >= 0 && w == 0) {
      tdm_load_tile((unsigned int)(uintptr_t)&ldsK[p ^ 1][0],
                    kbh + (size_t)(nxt * BLK) * (H_DIM * D_DIM));
      tdm_load_tile((unsigned int)(uintptr_t)&ldsVr[0],
                    vbh + (size_t)(nxt * BLK) * (H_DIM * D_DIM));
    }
    const _Float16* kcur = &ldsK[p][0];

    // ---- S = Q K^T : batch all 8 ds_load_b128 per n-tile, then WMMA chain ----
    v8f s[4];
#pragma unroll
    for (int nt = 0; nt < 4; ++nt) {
      v16h bfr[4];
#pragma unroll
      for (int kc = 0; kc < 4; ++kc) {
        const _Float16* bp = &kcur[(nt * 16 + ln) * KSTR + kc * 32 + koff8];
        bfr[kc] = load_frag(bp, bp + 16);
      }
      s[nt] = zero;
#pragma unroll
      for (int kc = 0; kc < 4; ++kc)
        s[nt] = __builtin_amdgcn_wmma_f32_16x16x32_f16(
            false, qf[kc], false, bfr[kc], (short)0, s[nt], false, false);
    }

    // ---- scale + diagonal causal mask + row max (shfl within 16 lanes) ----
    float rmax[8];
#pragma unroll
    for (int r = 0; r < 8; ++r) {
      int rs = hi ? (r + 8) : r;
#pragma unroll
      for (int nt = 0; nt < 4; ++nt) {
        float val = s[nt][r] * scale;
        if (diag && (nt * 16 + ln) > (w * 16 + rs)) val = -__builtin_inff();
        s[nt][r] = val;
      }
      float mx = fmaxf(fmaxf(s[0][r], s[1][r]), fmaxf(s[2][r], s[3][r]));
#pragma unroll
      for (int msk = 1; msk <= 8; msk <<= 1)
        mx = fmaxf(mx, __shfl_xor(mx, msk, 32));
      rmax[r] = mx;
    }

    // ---- online softmax update ----
#pragma unroll
    for (int r = 0; r < 8; ++r) {
      float mn    = fmaxf(m_i[r], rmax[r]);
      float alpha = __expf(m_i[r] - mn);
      m_i[r] = mn;
      float rsum = 0.0f;
#pragma unroll
      for (int nt = 0; nt < 4; ++nt) {
        float pe = __expf(s[nt][r] - mn);        // masked -inf -> 0
        s[nt][r] = pe;                           // reuse s[] as P
        rsum += pe;
      }
#pragma unroll
      for (int msk = 1; msk <= 8; msk <<= 1)
        rsum += __shfl_xor(rsum, msk, 32);
      l_i[r] = l_i[r] * alpha + rsum;
#pragma unroll
      for (int j = 0; j < 8; ++j) o[j][r] *= alpha;
    }

    // ---- spill P (f16) to this wave's private LDS tile ----
#pragma unroll
    for (int nt = 0; nt < 4; ++nt)
#pragma unroll
      for (int r = 0; r < 8; ++r) {
        int rs = hi ? (r + 8) : r;
        ldsP[(w * 16 + rs) * PSTR + nt * 16 + ln] = (_Float16)s[nt][r];
      }
    asm volatile("s_wait_dscnt 0" ::: "memory"); // store->load, same wave

    // ---- O += P V : batch loads per d-tile, then WMMA chain ----
#pragma unroll
    for (int j = 0; j < 8; ++j) {
      v16h af[2], bf2[2];
#pragma unroll
      for (int kc = 0; kc < 2; ++kc) {
        const _Float16* ap = &ldsP[(w * 16 + ln) * PSTR + kc * 32 + koff8];
        af[kc] = load_frag(ap, ap + 16);
        const _Float16* bp = &ldsVt[(j * 16 + ln) * VSTR + kc * 32 + koff8];
        bf2[kc] = load_frag(bp, bp + 16);
      }
#pragma unroll
      for (int kc = 0; kc < 2; ++kc)
        o[j] = __builtin_amdgcn_wmma_f32_16x16x32_f16(
            false, af[kc], false, bf2[kc], (short)0, o[j], false, false);
    }

    cur = nxt;
    p ^= 1;
  }

  // ---- epilogue: O / l, fp16 store ----
#pragma unroll
  for (int j = 0; j < 8; ++j) {
#pragma unroll
    for (int r = 0; r < 8; ++r) {
      int rs = hi ? (r + 8) : r;
      size_t off =
          ((((size_t)b * T_DIM) + (size_t)(qb * BLK + w * 16 + rs)) * H_DIM + h) * D_DIM
          + j * 16 + ln;
      out[off] = (_Float16)(o[j][r] / l_i[r]);
    }
  }
}

extern "C" void kernel_launch(void* const* d_in, const int* in_sizes, int n_in,
                              void* d_out, int out_size, void* d_ws, size_t ws_size,
                              hipStream_t stream) {
  const _Float16*      q  = (const _Float16*)d_in[0];
  const _Float16*      k  = (const _Float16*)d_in[1];
  const _Float16*      v  = (const _Float16*)d_in[2];
  const unsigned char* bm = (const unsigned char*)d_in[3];
  _Float16*            o  = (_Float16*)d_out;

  const int B = in_sizes[0] / (T_DIM * H_DIM * D_DIM);
  dim3 grid(NQ, H_DIM, B);
  bsattn_fwd_kernel<<<grid, 128, 0, stream>>>(q, k, v, bm, o);
}
